// VectorQuantizer_4063039062113
// MI455X (gfx1250) — matrix operand order repaired and checked by reading, not compiled, not run on hardware
//
#include <hip/hip_runtime.h>
#include <math.h>

#define N_TOKENS        65536
#define NUM_EMB         1024
#define EMB_DIM         512
#define TOK_PER_WAVE    32          // two 16-row A panels per wave (B reuse x2)
#define WAVES_PER_BLK   8
#define CODE_TILES      64          // 1024 codes / 16 per tile
#define TILE_BYTES      (16 * EMB_DIM * 2)   // 16 codes x 512 bf16 = 16 KB

typedef __attribute__((ext_vector_type(16))) __bf16 v16bf;
typedef __attribute__((ext_vector_type(8)))  float  v8f;
typedef unsigned int u32x4 __attribute__((ext_vector_type(4)));
typedef int          i32x8 __attribute__((ext_vector_type(8)));
typedef int          i32x4 __attribute__((ext_vector_type(4)));

#if defined(__has_builtin)
#if __has_builtin(__builtin_amdgcn_tensor_load_to_lds) && __has_builtin(__builtin_amdgcn_s_wait_tensorcnt)
#define USE_TDM 1
#endif
#endif
#ifndef USE_TDM
#define USE_TDM 0
#endif

// ---------------------------------------------------------------------------
// Kernel 0: convert codebook to bf16, compute ||w_k||^2, zero accumulators.
// ---------------------------------------------------------------------------
__global__ __launch_bounds__(128) void vq_prep(const float* __restrict__ w,
                                               __bf16* __restrict__ wbf,
                                               float* __restrict__ wnorm,
                                               float* __restrict__ counts,
                                               float* __restrict__ loss_sum) {
    const int row = blockIdx.x;          // one codebook row per block
    const int tid = threadIdx.x;         // 128 threads = 4 waves
    float acc = 0.f;
    for (int d = tid; d < EMB_DIM; d += 128) {
        float v = w[(size_t)row * EMB_DIM + d];
        wbf[(size_t)row * EMB_DIM + d] = (__bf16)v;
        acc += v * v;
    }
#pragma unroll
    for (int off = 16; off >= 1; off >>= 1) acc += __shfl_xor(acc, off, 32);
    __shared__ float s[4];
    if ((tid & 31) == 0) s[tid >> 5] = acc;
    __syncthreads();
    if (tid == 0) wnorm[row] = s[0] + s[1] + s[2] + s[3];
    if (row == 0) {                       // re-zero per launch: deterministic
        for (int i = tid; i < NUM_EMB; i += 128) counts[i] = 0.f;
        if (tid == 0) loss_sum[0] = 0.f;
    }
}

// ---------------------------------------------------------------------------
// TDM issue: DMA one 16-row x 1KB codebook tile (16 KB) global -> LDS.
// D# per CDNA5 ISA ch.8: 2D tile, data_size=3 (8B units), row = 128 units,
// tile 128x16, contiguous stride 128, descriptor type=2, count=1 valid.
// ---------------------------------------------------------------------------
#if USE_TDM
__device__ __forceinline__ void tdm_issue_tile(const __bf16* gsrc, unsigned lds_off) {
    const unsigned long long ga = (unsigned long long)(uintptr_t)gsrc;
    u32x4 g0;
    g0[0] = 1u;                                              // count=1, user mode
    g0[1] = lds_off;                                         // lds_addr [63:32]
    g0[2] = (unsigned)(ga & 0xffffffffu);                    // global_addr[31:0]
    g0[3] = (unsigned)((ga >> 32) & 0x1ffffffu) | (2u << 30);// addr[56:32] | type=2
    // d0: data_size=3 (8B) at [17:16]; no multicast/pad/iterate.
    // d1[31:16]=tensor_dim0 lo16 (128); d2[31:16]=tensor_dim1 lo16 (16)
    // d3[31:16]=tile_dim0 (128); d4[15:0]=tile_dim1 (16); d5=dim0_stride (128)
    i32x8 g1 = { 0x00030000, 0x00800000, 0x00100000, 0x00800000, 16, 128, 0, 0 };
    i32x4 gz4 = { 0, 0, 0, 0 };
    i32x8 gz8 = { 0, 0, 0, 0, 0, 0, 0, 0 };
    __builtin_amdgcn_tensor_load_to_lds(g0, g1, gz4, gz4, gz8, 0);
}
#endif

// ---------------------------------------------------------------------------
// Kernel 1: WMMA distance matmul + argmin + gather + loss partial sums.
// One wave = 32 tokens (two A panels resident, 256 VGPRs) so every 1KB B
// fragment fetched from LDS feeds TWO wmmas (halves LDS read traffic).
// 8 waves per block share each 16 KB codebook tile in LDS, double-buffered
// via the Tensor Data Mover.
// ---------------------------------------------------------------------------
__global__ __launch_bounds__(256) void vq_main(const float*  __restrict__ z,
                                               const float*  __restrict__ w,
                                               const __bf16* __restrict__ wbf,
                                               const float*  __restrict__ wnorm,
                                               float* __restrict__ counts,
                                               float* __restrict__ loss_sum,
                                               float* __restrict__ out) {
    const int tid   = threadIdx.x;
    const int lane  = tid & 31;
    const int wv    = tid >> 5;                        // wave in block (0..7)
    const int tile  = blockIdx.x * WAVES_PER_BLK + wv; // 32-token tile (0..2047)
    const int half  = lane >> 4;                       // 0: lanes 0-15, 1: 16-31
    const int nlane = lane & 15;

    __shared__ __align__(16) __bf16 s_b[2][16 * EMB_DIM];  // 2 x 16 KB B tiles
    __shared__ int s_best[WAVES_PER_BLK][TOK_PER_WAVE];

    // ---- Load two A panels (rows 0-15 and 16-31 of this 32-token tile).
    // A 16x32 layout (matches B): lanes 0-15 hold K {0..15},
    // lanes 16-31 hold K {16..31} for the same row M.
    const float* zr0 = z + (size_t)(tile * TOK_PER_WAVE + nlane) * EMB_DIM;
    const float* zr1 = zr0 + 16 * EMB_DIM;
    v16bf a0[16], a1[16];
#pragma unroll
    for (int c = 0; c < 16; ++c) {
        v8f lo0 = *(const v8f*)(zr0 + c * 32 + half * 16);
        v8f hi0 = *(const v8f*)(zr0 + c * 32 + 8 + half * 16);
        v8f lo1 = *(const v8f*)(zr1 + c * 32 + half * 16);
        v8f hi1 = *(const v8f*)(zr1 + c * 32 + 8 + half * 16);
        v16bf t0, t1;
#pragma unroll
        for (int i = 0; i < 8; ++i) {
            t0[i] = (__bf16)lo0[i]; t0[i + 8] = (__bf16)hi0[i];
            t1[i] = (__bf16)lo1[i]; t1[i + 8] = (__bf16)hi1[i];
        }
        a0[c] = t0; a1[c] = t1;
    }

    float mind0[8], mind1[8];
    int   midx0[8], midx1[8];
#pragma unroll
    for (int r = 0; r < 8; ++r) {
        mind0[r] = 3.4e38f; midx0[r] = 0;
        mind1[r] = 3.4e38f; midx1[r] = 0;
    }

#if USE_TDM
    if (wv == 0) {                                  // prime: DMA tile 0 -> buf 0
        tdm_issue_tile(wbf, (unsigned)(uintptr_t)&s_b[0][0]);
        __builtin_amdgcn_s_wait_tensorcnt(0);
    }
    __syncthreads();
#endif

    // ---- Sweep 64 codebook tiles of 16 codes each.
    for (int ct = 0; ct < CODE_TILES; ++ct) {
#if USE_TDM
        const int cur = ct & 1;
        if (wv == 0 && ct + 1 < CODE_TILES)         // overlap: DMA next tile
            tdm_issue_tile(wbf + (size_t)(ct + 1) * 16 * EMB_DIM,
                           (unsigned)(uintptr_t)&s_b[cur ^ 1][0]);
#else
        const int cur = 0;
        __syncthreads();                            // previous tile consumed
        {
            const float4* src = (const float4*)(wbf + (size_t)ct * 16 * EMB_DIM);
            float4* dst = (float4*)&s_b[0][0];
            for (int i = tid; i < TILE_BYTES / 16; i += 256) dst[i] = src[i];
        }
        __syncthreads();
#endif
        const int   n  = ct * 16 + nlane;           // this lane's code column
        const float wn = wnorm[n];
        // B layout (32x16 bf16): lane = column N; halves hold 16 consecutive K.
        const __bf16* wr = &s_b[cur][(size_t)nlane * EMB_DIM + half * 16];
        v8f acc0 = {}, acc1 = {};
#pragma unroll
        for (int c = 0; c < 16; ++c) {
            v16bf b = *(const v16bf*)(wr + c * 32); // ds_load_b128 x2, used twice
            acc0 = __builtin_amdgcn_wmma_f32_16x16x32_bf16(
                false, a0[c], false, b, (short)0, acc0, false, false);
            acc1 = __builtin_amdgcn_wmma_f32_16x16x32_bf16(
                false, a1[c], false, b, (short)0, acc1, false, false);
        }
        // dist = ||w||^2 - 2<z,w>  (||z||^2 constant per row -> irrelevant)
#pragma unroll
        for (int r = 0; r < 8; ++r) {
            float d0 = wn - 2.0f * acc0[r];
            float d1 = wn - 2.0f * acc1[r];
            if (d0 < mind0[r]) { mind0[r] = d0; midx0[r] = n; }
            if (d1 < mind1[r]) { mind1[r] = d1; midx1[r] = n; }
        }
#if USE_TDM
        if (wv == 0) __builtin_amdgcn_s_wait_tensorcnt(0); // next tile landed
        __syncthreads();                                   // publish to all waves
#endif
    }

    // ---- Reduce argmin across the 16 lanes sharing each row (both panels).
#pragma unroll
    for (int r = 0; r < 8; ++r) {
        float m0 = mind0[r], m1 = mind1[r];
        int   i0 = midx0[r], i1 = midx1[r];
#pragma unroll
        for (int off = 8; off >= 1; off >>= 1) {    // xor<16 stays in 16-lane group
            float o0 = __shfl_xor(m0, off, 32);
            int   q0 = __shfl_xor(i0, off, 32);
            float o1 = __shfl_xor(m1, off, 32);
            int   q1 = __shfl_xor(i1, off, 32);
            if (o0 < m0 || (o0 == m0 && q0 < i0)) { m0 = o0; i0 = q0; }
            if (o1 < m1 || (o1 == m1 && q1 < i1)) { m1 = o1; i1 = q1; }
        }
        if (nlane == 0) {                           // lanes 0 & 16: rows r, r+8
            s_best[wv][r + 8 * half]      = i0;     // panel 0: rows 0-15
            s_best[wv][16 + r + 8 * half] = i1;     // panel 1: rows 16-31
            atomicAdd(&counts[i0], 1.0f);
            atomicAdd(&counts[i1], 1.0f);
        }
    }
    __syncthreads();

    // ---- Gather z_q = w[best] (f32), write output, accumulate ||z_q - z||^2.
    float acc2 = 0.f;
    const float4* zf4 = (const float4*)z;
    const float4* wf4 = (const float4*)w;
    float4*       of4 = (float4*)out;
    for (int j = 0; j < TOK_PER_WAVE; ++j) {
        const int    best = s_best[wv][j];
        const size_t row  = (size_t)tile * TOK_PER_WAVE + j;
#pragma unroll
        for (int it = 0; it < 4; ++it) {            // 128 float4 per row / 32 lanes
            const int d4 = lane + it * 32;
            float4 q = wf4[(size_t)best * 128 + d4];
            float4 v = zf4[row * 128 + d4];
            of4[row * 128 + d4] = q;
            float dx = q.x - v.x, dy = q.y - v.y, dz = q.z - v.z, dw = q.w - v.w;
            acc2 += dx * dx + dy * dy + dz * dz + dw * dw;
        }
    }
#pragma unroll
    for (int off = 16; off >= 1; off >>= 1) acc2 += __shfl_xor(acc2, off, 32);
    if (lane == 0) atomicAdd(loss_sum, acc2);
}

// ---------------------------------------------------------------------------
// Kernel 2: loss scaling + perplexity from histogram.
// ---------------------------------------------------------------------------
__global__ __launch_bounds__(256) void vq_finalize(const float* __restrict__ counts,
                                                   const float* __restrict__ loss_sum,
                                                   float* __restrict__ out_tail) {
    const int tid = threadIdx.x;
    float local = 0.f;
    for (int i = tid; i < NUM_EMB; i += 256) {
        float e = counts[i] * (1.0f / (float)N_TOKENS);
        local += e * logf(e + 1e-10f);
    }
#pragma unroll
    for (int off = 16; off >= 1; off >>= 1) local += __shfl_xor(local, off, 32);
    __shared__ float s[8];
    if ((tid & 31) == 0) s[tid >> 5] = local;
    __syncthreads();
    if (tid == 0) {
        float t = 0.f;
        for (int i = 0; i < 8; ++i) t += s[i];
        out_tail[0] = loss_sum[0] * (1.25f / ((float)N_TOKENS * (float)EMB_DIM));
        out_tail[1] = expf(-t);
    }
}

// ---------------------------------------------------------------------------
extern "C" void kernel_launch(void* const* d_in, const int* in_sizes, int n_in,
                              void* d_out, int out_size, void* d_ws, size_t ws_size,
                              hipStream_t stream) {
    const float* z = (const float*)d_in[0];   // [65536, 512] f32
    const float* w = (const float*)d_in[1];   // [1024, 512]  f32
    float* out = (float*)d_out;               // z_q_st flat, then loss, perplexity

    char*   ws       = (char*)d_ws;
    __bf16* wbf      = (__bf16*)ws;                                  // 1 MB
    float*  wnorm    = (float*)(ws + (1u << 20));                    // 4 KB
    float*  counts   = (float*)(ws + (1u << 20) + 4096);             // 4 KB
    float*  loss_sum = (float*)(ws + (1u << 20) + 8192);             // 4 B

    vq_prep<<<NUM_EMB, 128, 0, stream>>>(w, wbf, wnorm, counts, loss_sum);

    const int nblocks = N_TOKENS / (TOK_PER_WAVE * WAVES_PER_BLK);   // 256
    vq_main<<<nblocks, 256, 0, stream>>>(z, w, wbf, wnorm, counts, loss_sum, out);

    vq_finalize<<<1, 256, 0, stream>>>(counts, loss_sum,
                                       out + (size_t)N_TOKENS * EMB_DIM);
}
